// HGN_AD_9543417332148
// MI455X (gfx1250) — matrix operand order repaired and checked by reading, not compiled, not run on hardware
//
#include <hip/hip_runtime.h>
#include <hip/hip_bf16.h>

// ---------------------------------------------------------------------------
// HGN attention layer for MI455X (gfx1250, wave32).
// - GEMMs on V_WMMA_F32_16X16X32_BF16 (bf16 in, f32 accumulate).
// - hbf is stored in WMMA-B fragment order by kernel 1, so kernel 3 stages it
//   with GLOBAL_LOAD_ASYNC_TO_LDS_B128 (ASYNCcnt) as a pure 64 KB linear copy
//   and runs its whole K-loop out of LDS with zero staging barriers.
// - Cross-lane reductions are single-instruction ds_swizzle XOR patterns.
// - y8 = a2^T @ feats and the pair edgeconv also run on the WMMA pipe.
// ---------------------------------------------------------------------------

typedef __attribute__((ext_vector_type(16))) __bf16 v16bf;
typedef __attribute__((ext_vector_type(8)))  float  v8f;

#define T_DIM 32
#define N_DIM 512
#define E_DIM 64
#define F_DIM 64
#define HD    64
#define TN    (T_DIM * N_DIM)
#define NEGV  (-1e9f)
#define SLOPE 0.2f

// 16-bit A/B fragment element->K map used here: contiguous, K = j + 16*half.
__device__ __forceinline__ int a_kmap(int j, int half) {
  return j + 16 * half;
}
__device__ __forceinline__ float leaky(float v) { return v > 0.f ? v : SLOPE * v; }
__device__ __forceinline__ float eluf(float v)  { return v > 0.f ? v : __expf(v) - 1.f; }

__device__ __forceinline__ v8f wmma_bf16(const v16bf& A, const v16bf& B, v8f C) {
  return __builtin_amdgcn_wmma_f32_16x16x32_bf16(false, A, false, B, (short)0, C,
                                                 false, false);
}

// ds_swizzle group-of-32 XOR patterns: offset = (xor_mask<<10) | and_mask(0x1f).
template <int PAT>
__device__ __forceinline__ float swz_add(float v) {
  return v + __builtin_bit_cast(float, __builtin_amdgcn_ds_swizzle(
                 __builtin_bit_cast(int, v), PAT));
}
template <int PAT>
__device__ __forceinline__ float swz_max(float v) {
  return fmaxf(v, __builtin_bit_cast(float, __builtin_amdgcn_ds_swizzle(
                      __builtin_bit_cast(int, v), PAT)));
}
__device__ __forceinline__ float red16_add(float v) {  // XOR 1,2,4,8 (stays in half)
  v = swz_add<0x041f>(v); v = swz_add<0x081f>(v);
  v = swz_add<0x101f>(v); v = swz_add<0x201f>(v);
  return v;
}
__device__ __forceinline__ float red32_add(float v) { return swz_add<0x401f>(red16_add(v)); }
__device__ __forceinline__ float red32_max(float v) {
  v = swz_max<0x041f>(v); v = swz_max<0x081f>(v); v = swz_max<0x101f>(v);
  v = swz_max<0x201f>(v); v = swz_max<0x401f>(v);
  return v;
}
__device__ __forceinline__ float bcast0(float v) {
  return __builtin_bit_cast(float, __builtin_amdgcn_readfirstlane(
             __builtin_bit_cast(int, v)));
}
__device__ __forceinline__ unsigned int pack_bf16(float a, float b) {
  return (unsigned int)__builtin_bit_cast(unsigned short, (__bf16)a) |
         ((unsigned int)__builtin_bit_cast(unsigned short, (__bf16)b) << 16);
}

// Swizzle a 64x64 f32 matrix into WMMA-B fragments in LDS (one slot per thread).
__device__ __forceinline__ void stage_b64x64(const float* __restrict__ src,
                                             v16bf* __restrict__ dst, int tid) {
  const int ti = tid >> 5, ln = tid & 31;
  const int kc = ti >> 2, nt = ti & 3;
  const int hf = ln >> 4, lm = ln & 15;
  v16bf v;
#pragma unroll
  for (int j = 0; j < 16; ++j)
    v[j] = (__bf16)src[(32 * kc + j + 16 * hf) * HD + nt * 16 + lm];
  dst[ti * 32 + ln] = v;
}

// ---------------------------------------------------------------------------
// Kernel 0: bit-pack H/adj, and precompute wa = W @ a (since s = h.a = x.(W a)).
// ---------------------------------------------------------------------------
__global__ void __launch_bounds__(256)
k_pack(const int* __restrict__ H, const int* __restrict__ adj,
       const float* __restrict__ W, const float* __restrict__ a_hyp,
       const float* __restrict__ a_ind, unsigned int* __restrict__ Hm,
       unsigned int* __restrict__ adjm, float* __restrict__ wa) {
  const int gid = blockIdx.x * 256 + threadIdx.x;
  if (gid < N_DIM * 16) {  // adj: 8192 dwords
    const int row = gid >> 4, w = gid & 15;
    unsigned int b = 0;
    for (int i = 0; i < 32; ++i)
      b |= adj[row * N_DIM + w * 32 + i] ? (1u << i) : 0u;
    adjm[gid] = b;
  }
  if (gid < N_DIM * 2) {   // H: 1024 dwords
    const int row = gid >> 1, w = gid & 1;
    unsigned int b = 0;
    for (int i = 0; i < 32; ++i)
      b |= H[row * E_DIM + w * 32 + i] ? (1u << i) : 0u;
    Hm[gid] = b;
  }
  if (gid < 2 * F_DIM) {   // wa[0:64]=W@a_hyp, wa[64:128]=W@a_ind
    const int which = gid >> 6, f = gid & 63;
    const float* a = which ? a_ind : a_hyp;
    float s = 0.f;
    for (int d = 0; d < HD; ++d) s += W[f * HD + d] * a[d];
    wa[gid] = s;
  }
}

// ---------------------------------------------------------------------------
// Kernel 1: h = x @ W. Scores folded into the A-tile build (one swizzle each).
// hbfB is written directly in WMMA-B fragment order: the C/D epilogue registers
// acc[nt][0..7] are 8 consecutive fragment elements -> one b128 store per nt,
// and lanes cover a fully contiguous 512B span (perfect coalescing).
//   hbfB element addr = ((chunk*4 + nt)*32 + lm + 16*hfp)*16 + (r + 8*half)
//   chunk = row/32, hfp = (row/16)&1.
// ---------------------------------------------------------------------------
__global__ void __launch_bounds__(256)
k_h_gemm(const float* __restrict__ x, const float* __restrict__ W,
         const float* __restrict__ wa, float* __restrict__ h,
         __bf16* __restrict__ hbfB, float* __restrict__ s_hyp,
         float* __restrict__ s_ind) {
  __shared__ v16bf lWB[256];
  __shared__ float lwa[2 * F_DIM];
  const int tid = threadIdx.x;
  stage_b64x64(W, lWB, tid);
  if (tid < 2 * F_DIM) lwa[tid] = wa[tid];
  __syncthreads();

  const int wave = tid >> 5, lane = tid & 31, half = lane >> 4, lm = lane & 15;
  const int rbase = blockIdx.x * 128 + wave * 16;

  v16bf Breg[8];
#pragma unroll
  for (int ti = 0; ti < 8; ++ti) Breg[ti] = lWB[ti * 32 + lane];

  v8f acc[4];
#pragma unroll
  for (int nt = 0; nt < 4; ++nt)
#pragma unroll
    for (int r = 0; r < 8; ++r) acc[nt][r] = 0.f;

  float ph = 0.f, pi = 0.f;  // partial x . (W a) for row lm (this lane's half)
#pragma unroll
  for (int kc = 0; kc < 2; ++kc) {
    v16bf A;
#pragma unroll
    for (int j = 0; j < 16; ++j) {
      const int K = 32 * kc + a_kmap(j, half);
      const float f = x[(rbase + lm) * F_DIM + K];
      A[j] = (__bf16)f;
      ph += f * lwa[K];
      pi += f * lwa[F_DIM + K];
    }
#pragma unroll
    for (int nt = 0; nt < 4; ++nt)
      acc[nt] = wmma_bf16(A, Breg[kc * 4 + nt], acc[nt]);
  }
  ph = swz_add<0x401f>(ph);  // combine the two halves of row lm
  pi = swz_add<0x401f>(pi);
  if (half == 0) {
    s_hyp[rbase + lm] = leaky(ph);
    s_ind[rbase + lm] = leaky(pi);
  }

  const int chunk = rbase >> 5;        // 32-row chunk id
  const int hfp   = (rbase >> 4) & 1;  // which 16-row half of the chunk
#pragma unroll
  for (int nt = 0; nt < 4; ++nt) {
    // f32 h (row-major, consumed per-row by kernel 4)
#pragma unroll
    for (int r = 0; r < 8; ++r)
      h[(rbase + r + 8 * half) * HD + nt * 16 + lm] = acc[nt][r];
    // bf16 h in B-fragment order: one b128 store
    uint4 pk;
    pk.x = pack_bf16(acc[nt][0], acc[nt][1]);
    pk.y = pack_bf16(acc[nt][2], acc[nt][3]);
    pk.z = pack_bf16(acc[nt][4], acc[nt][5]);
    pk.w = pack_bf16(acc[nt][6], acc[nt][7]);
    const size_t eoff = ((size_t)((chunk * 4 + nt) * 32 + lm + 16 * hfp)) * 16 + 8 * half;
    *reinterpret_cast<uint4*>(hbfB + eoff) = pk;
  }
}

// ---------------------------------------------------------------------------
// Kernel 2: hyperedge denominators iz[t,e] = 1 / sum_{n in e} exp(s_hyp[t,n]).
// ---------------------------------------------------------------------------
__global__ void __launch_bounds__(256)
k_hyper_stats(const unsigned int* __restrict__ Hm, const float* __restrict__ s_hyp,
              float* __restrict__ iz_ws) {
  const int tid = threadIdx.x, wave = tid >> 5, lane = tid & 31;
  const int pair = blockIdx.x * 8 + wave;
  const int t = pair >> 6, e = pair & 63;
  const unsigned int ebit = 1u << (e & 31);
  const int ew = e >> 5;

  float s = 0.f;
  for (int it = 0; it < 16; ++it) {
    const int n = it * 32 + lane;
    if (Hm[n * 2 + ew] & ebit) s += __expf(s_hyp[t * N_DIM + n]);
  }
  s = red32_add(s);
  if (lane == 0) iz_ws[t * E_DIM + e] = (s > 0.f) ? 1.f / s : 0.f;
}

// ---------------------------------------------------------------------------
// Kernel 3: industry[t] = row_softmax(mask(adj, s_ind[t])) @ h[t]
// Stages all 16 pre-swizzled K-chunks of h[t] (64 KB) into LDS with async-to-
// LDS b128 copies, overlapped with the masked row-sum pass; the WMMA K-loop
// then runs barrier-free entirely out of LDS.
// ---------------------------------------------------------------------------
__global__ void __launch_bounds__(256)
k_industry(const unsigned int* __restrict__ adjm, const float* __restrict__ s_ind,
           const __bf16* __restrict__ hbfB, float* __restrict__ industry) {
  __shared__ float les[N_DIM];       // exp(s_ind[t, :])
  __shared__ v16bf lhAll[16 * 128];  // all K-chunks, B-fragment order (64 KB)
  const int tid = threadIdx.x, wave = tid >> 5, lane = tid & 31;
  const int half = lane >> 4, lm = lane & 15;
  const int t = blockIdx.x >> 2;
  const int ibase = (blockIdx.x & 3) * 128 + wave * 16;
  const int row = ibase + lm;

  // Kick off the async staging volley: 16 x b128 per thread (ASYNCcnt-tracked).
  {
    const unsigned lds0 =
        (unsigned)(size_t)&lhAll[0] + (unsigned)tid * 16u;
    const unsigned long long g0 =
        (unsigned long long)(const void*)hbfB + (size_t)t * 65536 + (size_t)tid * 16;
#pragma unroll
    for (int kc = 0; kc < 16; ++kc) {
      asm volatile("global_load_async_to_lds_b128 %0, %1, off"
                   :: "v"(lds0 + (unsigned)kc * 4096u),
                      "v"(g0 + (unsigned long long)kc * 4096ull)
                   : "memory");
    }
  }

  for (int i = tid; i < N_DIM; i += 256) les[i] = __expf(s_ind[t * N_DIM + i]);
  unsigned int madj[16];
#pragma unroll
  for (int w = 0; w < 16; ++w) madj[w] = adjm[row * 16 + w];
  __syncthreads();  // les visible (async copies still in flight)

  // Row denominator: iterate only over set bits; halves split the j-range.
  float sum = 0.f;
#pragma unroll 1
  for (int w = 0; w < 8; ++w) {
    unsigned int bits = madj[8 * half + w];
    while (bits) {
      const int b = __builtin_ctz(bits);
      bits &= bits - 1u;
      sum += les[(8 * half + w) * 32 + b];
    }
  }
  sum = swz_add<0x401f>(sum);
  const float iz = 1.f / sum;  // diagonal guarantees sum > 0

  v8f acc[4];
#pragma unroll
  for (int nt = 0; nt < 4; ++nt)
#pragma unroll
    for (int r = 0; r < 8; ++r) acc[nt][r] = 0.f;

  asm volatile("s_wait_asynccnt 0x0" ::: "memory");  // this wave's copies done
  __syncthreads();                                   // all waves' copies done

#pragma unroll 1
  for (int kc = 0; kc < 16; ++kc) {
    v16bf A;  // beta tile from register bitmask + staged exp values
#pragma unroll
    for (int j = 0; j < 16; ++j) {
      const int jg = kc * 32 + a_kmap(j, half);
      const int bit = (madj[jg >> 5] >> (jg & 31)) & 1;
      A[j] = bit ? (__bf16)(les[jg] * iz) : (__bf16)0.f;
    }
#pragma unroll
    for (int nt = 0; nt < 4; ++nt)
      acc[nt] = wmma_bf16(A, lhAll[kc * 128 + nt * 32 + lane], acc[nt]);
  }

#pragma unroll
  for (int r = 0; r < 8; ++r) {
    const int ri = ibase + r + 8 * half;
#pragma unroll
    for (int nt = 0; nt < 4; ++nt)
      industry[(size_t)(t * N_DIM + ri) * HD + nt * 16 + lm] = acc[nt][r];
  }
}

// ---------------------------------------------------------------------------
// Kernel 4 (dominant): fused per-(t,n) edgeconv; one wave per (t,n).
// 24 static WMMAs; Wc1's 8 B fragments resident in VGPRs; one block barrier.
// ---------------------------------------------------------------------------
__global__ void __launch_bounds__(256)
k_edgeconv_fused(const unsigned int* __restrict__ Hm, const float* __restrict__ h,
                 const float* __restrict__ s_hyp, const float* __restrict__ iz_ws,
                 const float* __restrict__ industry, const float* __restrict__ Wc1,
                 const float* __restrict__ bc1, const float* __restrict__ wc2,
                 float* __restrict__ out) {
  __shared__ v16bf lWc1B[256];
  __shared__ float lbc1[HD], lwc2[HD], liz[E_DIM];
  __shared__ float lh[8][HD], lal[8][E_DIM], lz[8][E_DIM];

  const int tid = threadIdx.x, wave = tid >> 5, lane = tid & 31;
  const int half = lane >> 4, lm = lane & 15;
  const int gw = blockIdx.x * 8 + wave;
  const int t = gw >> 9, n = gw & 511;

  stage_b64x64(Wc1, lWc1B, tid);
  if (tid < HD) {
    lbc1[tid] = bc1[tid];
    lwc2[tid] = wc2[tid];
    liz[tid]  = iz_ws[(blockIdx.x >> 6) * E_DIM + tid];  // block-uniform t
  }

  const float sh  = s_hyp[t * N_DIM + n];
  const float esh = __expf(sh);
  const int d0 = 2 * lane, d1 = d0 + 1;
  const float2 hv = *reinterpret_cast<const float2*>(h + (size_t)(t * N_DIM + n) * HD + d0);
  *reinterpret_cast<float2*>(&lh[wave][d0]) = hv;
  const unsigned int hm = Hm[n * 2 + (d0 >> 5)];
  const int mk0 = (hm >> (d0 & 31)) & 1, mk1 = (hm >> (d1 & 31)) & 1;
  __builtin_prefetch(industry + (size_t)(t * N_DIM + n) * HD, 0, 1);
  __syncthreads();  // lWc1B/liz/lh visible block-wide

  {
    float2 al;
    al.x = mk0 ? esh * liz[d0] : 0.f;
    al.y = mk1 ? esh * liz[d1] : 0.f;
    *reinterpret_cast<float2*>(&lal[wave][d0]) = al;  // alpha row
  }
  __builtin_amdgcn_wave_barrier();

  v16bf Breg[8];
#pragma unroll
  for (int ti = 0; ti < 8; ++ti) Breg[ti] = lWc1B[ti * 32 + lane];

  float hreg[2][16];  // this lane's h values in A-fragment order
#pragma unroll
  for (int kc = 0; kc < 2; ++kc)
#pragma unroll
    for (int j = 0; j < 16; ++j)
      hreg[kc][j] = lh[wave][32 * kc + a_kmap(j, half)];

  // --- z[e] = relu(feats@Wc1 + bc1) @ wc2 ---
#pragma unroll 1
  for (int mt = 0; mt < 4; ++mt) {
    v8f acc[4];
#pragma unroll
    for (int nt = 0; nt < 4; ++nt)
#pragma unroll
      for (int r = 0; r < 8; ++r) acc[nt][r] = 0.f;

    const float ae = lal[wave][mt * 16 + lm];
#pragma unroll
    for (int kc = 0; kc < 2; ++kc) {
      v16bf A;
#pragma unroll
      for (int j = 0; j < 16; ++j)
        A[j] = (__bf16)eluf(ae * hreg[kc][j]);
#pragma unroll
      for (int nt = 0; nt < 4; ++nt)
        acc[nt] = wmma_bf16(A, Breg[kc * 4 + nt], acc[nt]);
    }
#pragma unroll
    for (int r = 0; r < 8; ++r) {
      float zp = 0.f;
#pragma unroll
      for (int nt = 0; nt < 4; ++nt) {
        const int dd = nt * 16 + lm;
        zp += fmaxf(acc[nt][r] + lbc1[dd], 0.f) * lwc2[dd];
      }
      zp = red16_add(zp);
      if (lm == 0) lz[wave][mt * 16 + r + 8 * half] = zp;
    }
  }
  __builtin_amdgcn_wave_barrier();

  // --- masked softmax over 64 hyperedges (each lane owns two e's) ---
  const float z0 = mk0 ? lz[wave][d0] : NEGV;
  const float z1 = mk1 ? lz[wave][d1] : NEGV;
  const float mz = red32_max(fmaxf(z0, z1));
  const float ez0 = __expf(z0 - mz), ez1 = __expf(z1 - mz);
  const float inv = 1.f / red32_add(ez0 + ez1);
  {
    float2 a2; a2.x = ez0 * inv; a2.y = ez1 * inv;
    *reinterpret_cast<float2*>(&lz[wave][d0]) = a2;
  }
  __builtin_amdgcn_wave_barrier();

  // --- y8 = a2^T @ feats on the WMMA pipe (A padded to 16 rows, row 0 live) ---
  float hcol[4];
#pragma unroll
  for (int nt = 0; nt < 4; ++nt) hcol[nt] = lh[wave][nt * 16 + lm];
  float areg[2][16], a2A[2][16];
#pragma unroll
  for (int kc = 0; kc < 2; ++kc)
#pragma unroll
    for (int j = 0; j < 16; ++j) {
      areg[kc][j] = lal[wave][32 * kc + j + 16 * half];   // B-fragment k order
      a2A[kc][j]  = lz[wave][32 * kc + a_kmap(j, half)];  // A-fragment order
    }
  v8f accy[4];
#pragma unroll
  for (int nt = 0; nt < 4; ++nt)
#pragma unroll
    for (int r = 0; r < 8; ++r) accy[nt][r] = 0.f;
#pragma unroll
  for (int kc = 0; kc < 2; ++kc) {
    v16bf A3;
#pragma unroll
    for (int j = 0; j < 16; ++j)
      A3[j] = (lm == 0) ? (__bf16)a2A[kc][j] : (__bf16)0.f;
#pragma unroll
    for (int nt = 0; nt < 4; ++nt) {
      v16bf B3;  // feats fragment: B[k=e][n=d] = elu(alpha[e] * h[d])
#pragma unroll
      for (int j = 0; j < 16; ++j)
        B3[j] = (__bf16)eluf(areg[kc][j] * hcol[nt]);
      accy[nt] = wmma_bf16(A3, B3, accy[nt]);
    }
  }
  __builtin_amdgcn_wave_barrier();
  if (half == 0) {  // D row M=0 -> y8; scatter to LDS (overwrites h row)
#pragma unroll
    for (int nt = 0; nt < 4; ++nt) lh[wave][nt * 16 + lm] = accy[nt][0];
  }
  const float2 ind = *reinterpret_cast<const float2*>(
      industry + (size_t)(t * N_DIM + n) * HD + d0);
  *reinterpret_cast<float2*>(&lal[wave][d0]) = ind;  // feats2 row 0
  __builtin_amdgcn_wave_barrier();

  // --- pair edgeconv over {industry, y8}: padded 16x64 A-tile ---
  v8f acc2[4];
#pragma unroll
  for (int nt = 0; nt < 4; ++nt)
#pragma unroll
    for (int r = 0; r < 8; ++r) acc2[nt][r] = 0.f;
#pragma unroll
  for (int kc = 0; kc < 2; ++kc) {
    v16bf A2;
#pragma unroll
    for (int j = 0; j < 16; ++j) {
      const int d = 32 * kc + a_kmap(j, half);
      const float v = (lm == 0) ? lal[wave][d] : (lm == 1) ? lh[wave][d] : 0.f;
      A2[j] = (__bf16)v;
    }
#pragma unroll
    for (int nt = 0; nt < 4; ++nt)
      acc2[nt] = wmma_bf16(A2, Breg[kc * 4 + nt], acc2[nt]);
  }
  float zp0 = 0.f, zp1 = 0.f;  // rows M=0,1 live in half 0 (r=0,1)
#pragma unroll
  for (int nt = 0; nt < 4; ++nt) {
    const int dd = nt * 16 + lm;
    zp0 += fmaxf(acc2[nt][0] + lbc1[dd], 0.f) * lwc2[dd];
    zp1 += fmaxf(acc2[nt][1] + lbc1[dd], 0.f) * lwc2[dd];
  }
  const float z2_0 = bcast0(red16_add(zp0));
  const float z2_1 = bcast0(red16_add(zp1));

  const float2 y8 = *reinterpret_cast<const float2*>(&lh[wave][d0]);
  const float mm = fmaxf(z2_0, z2_1);
  const float w0 = __expf(z2_0 - mm), w1 = __expf(z2_1 - mm);
  const float wi = 1.f / (w0 + w1);
  float2 o;
  o.x = (w0 * ind.x + w1 * y8.x) * wi;
  o.y = (w0 * ind.y + w1 * y8.y) * wi;
  *reinterpret_cast<float2*>(out + (size_t)(t * N_DIM + n) * HD + d0) = o;
}

// ---------------------------------------------------------------------------
// Launch. Inputs: 0 x, 1 H, 2 adj, 3 nhid, 4 W, 5 a_hyp, 6 a_ind, 7 Wc1,
// 8 bc1, 9 wc2, 10 bc2 (bc2 cancels inside softmax). Workspace (bytes):
//   h        f32 [TN,64]  @ 0          | industry f32 [TN,64] @ 6,430,720
//   hbfB     bf16[TN,64]  @ 4,194,304  | Hm  u32 [N,2]   @ 10,625,024
//   s_hyp    f32 [TN]     @ 6,291,456  | adjm u32 [N,16] @ 10,629,120
//   s_ind    f32 [TN]     @ 6,356,992  | wa  f32 [2,64]  @ 10,661,888
//   iz_ws    f32 [T,E]    @ 6,422,528  | total ~10.66 MB
// ---------------------------------------------------------------------------
extern "C" void kernel_launch(void* const* d_in, const int* in_sizes, int n_in,
                              void* d_out, int out_size, void* d_ws,
                              size_t ws_size, hipStream_t stream) {
  const float* x     = (const float*)d_in[0];
  const int*   H     = (const int*)d_in[1];
  const int*   adj   = (const int*)d_in[2];
  const float* W     = (const float*)d_in[4];
  const float* a_hyp = (const float*)d_in[5];
  const float* a_ind = (const float*)d_in[6];
  const float* Wc1   = (const float*)d_in[7];
  const float* bc1   = (const float*)d_in[8];
  const float* wc2   = (const float*)d_in[9];

  char*         ws       = (char*)d_ws;
  float*        h        = (float*)(ws + 0);
  __bf16*       hbfB     = (__bf16*)(ws + 4194304);
  float*        s_hyp    = (float*)(ws + 6291456);
  float*        s_ind    = (float*)(ws + 6356992);
  float*        iz_ws    = (float*)(ws + 6422528);
  float*        industry = (float*)(ws + 6430720);
  unsigned int* Hm       = (unsigned int*)(ws + 10625024);
  unsigned int* adjm     = (unsigned int*)(ws + 10629120);
  float*        wa       = (float*)(ws + 10661888);
  float*        out      = (float*)d_out;

  k_pack<<<32, 256, 0, stream>>>(H, adj, W, a_hyp, a_ind, Hm, adjm, wa);
  k_h_gemm<<<TN / 128, 256, 0, stream>>>(x, W, wa, h, hbfB, s_hyp, s_ind);
  k_hyper_stats<<<(T_DIM * E_DIM) / 8, 256, 0, stream>>>(Hm, s_hyp, iz_ws);
  k_industry<<<T_DIM * (N_DIM / 128), 256, 0, stream>>>(adjm, s_ind, hbfB, industry);
  k_edgeconv_fused<<<TN / 8, 256, 0, stream>>>(Hm, h, s_hyp, iz_ws, industry,
                                               Wc1, bc1, wc2, out);
}
